// MaskedMultiHeadAttention_23536420782867
// MI455X (gfx1250) — compile-verified, matrix operand order checked by reading, not run on hardware
//
#include <hip/hip_runtime.h>
#include <hip/hip_bf16.h>
#include <math.h>

// ---------------- problem constants ----------------
constexpr int CB  = 4;      // batch
constexpr int CS  = 2048;   // sequence
constexpr int CDM = 1024;   // model dim
constexpr int CH  = 16;     // heads
constexpr int CHD = 64;     // head dim
constexpr int LDSP = 40;    // padded LDS row stride (bf16 elems): 80B = 20 banks

typedef __attribute__((ext_vector_type(16))) __bf16 v16bf;
typedef __attribute__((ext_vector_type(8)))  float  v8f;
typedef __attribute__((ext_vector_type(4)))  unsigned int u32x4;
typedef __attribute__((ext_vector_type(8)))  int i32x8;
typedef __attribute__((ext_vector_type(4)))  int i32x4;

#if __has_builtin(__builtin_amdgcn_tensor_load_to_lds) && \
    __has_builtin(__builtin_amdgcn_s_wait_tensorcnt)
#define USE_TDM 1
#else
#define USE_TDM 0
#endif

// ---------------- WMMA helpers ----------------
__device__ __forceinline__ v8f wmma_bf16(v16bf a, v16bf b, v8f c) {
  // D(f32 16x16) = A(16x32 bf16) * B(32x16 bf16) + C
  return __builtin_amdgcn_wmma_f32_16x16x32_bf16(false, a, false, b, (short)0, c,
                                                 false, false);
}

__device__ __forceinline__ v8f v8f_zero() {
  v8f z = {0.f, 0.f, 0.f, 0.f, 0.f, 0.f, 0.f, 0.f};
  return z;
}

// Fragment with contiguous K runs. Works for BOTH:
//  - A fragment (16x32): rowBase = row `m` of A (K contiguous)
//  - B fragment (32x16): rowBase = column `n` of B stored K-contiguous
// Per ISA: lanes0-15 hold K=0..7 then 16..23; lanes16-31 hold 8..15 then 24..31
// => per lane two contiguous 8-elem (16B) segments at +8*hi and +16+8*hi.
__device__ __forceinline__ v16bf load_frag16(const __bf16* rowBase, int hi) {
  union { v16bf v; uint4 q[2]; } u;
  u.q[0] = *reinterpret_cast<const uint4*>(rowBase + 8 * hi);
  u.q[1] = *reinterpret_cast<const uint4*>(rowBase + 16 + 8 * hi);
  return u.v;
}

// Same layout, converting fp32 -> bf16 on the fly (for fp32 activations).
__device__ __forceinline__ v16bf load_frag16_f32(const float* rowBase, int hi) {
  union { v16bf v; __bf16 e[16]; } u;
  const float* p0 = rowBase + 8 * hi;
  const float* p1 = rowBase + 16 + 8 * hi;
#pragma unroll
  for (int i = 0; i < 8; ++i) {
    u.e[i]     = (__bf16)p0[i];
    u.e[i + 8] = (__bf16)p1[i];
  }
  return u.v;
}

#if USE_TDM
// Issue a TDM 2D tile load: Wt[row0..row0+tile1) x [col0..col0+tile0) bf16
// -> LDS at ldsByteAddr, with 16B padding after every 64B row (row stride
// becomes LDSP*2 = 80B). D# built per CDNA5 ISA sec. 8 (groups 0/1; 2D so
// groups 2/3 are zero).  6-arg builtin form (clang-23 / therock headers).
__device__ __forceinline__ void tdm_load_tile_2d(const __bf16* gTile,
                                                 unsigned ldsByteAddr,
                                                 unsigned tensorDim0,
                                                 unsigned tensorDim1,
                                                 unsigned tileDim0,
                                                 unsigned tileDim1,
                                                 unsigned rowStrideElems) {
  unsigned long long ga = (unsigned long long)gTile;
  u32x4 g0;
  g0[0] = 1u;                                   // count=1, user descriptor
  g0[1] = ldsByteAddr;                          // lds_addr
  g0[2] = (unsigned)(ga & 0xffffffffu);         // global_addr[31:0]
  g0[3] = (unsigned)((ga >> 32) & 0x01ffffffu)  // global_addr[56:32]
          | (2u << 30);                         // type = 2 ("image")
  i32x8 g1;
  // data_size=1(2B)<<16 | pad_enable<<20 | pad_interval=3 (64B)<<22 |
  // pad_amount=3 (4 dwords = 16B)<<25 ; workgroup_mask=0 (not in cluster)
  g1[0] = (1 << 16) | (1 << 20) | (3 << 22) | (3 << 25);
  g1[1] = (int)((tensorDim0 & 0xffffu) << 16);                    // dim0 lo16
  g1[2] = (int)((tensorDim0 >> 16) | ((tensorDim1 & 0xffffu) << 16));
  g1[3] = (int)((tensorDim1 >> 16) | (tileDim0 << 16));
  g1[4] = (int)(tileDim1 & 0xffffu);            // tile_dim1; tile_dim2=0
  g1[5] = (int)rowStrideElems;                  // tensor_dim0_stride lo32
  g1[6] = 0;
  g1[7] = 0;
  i32x4 zz4 = {0, 0, 0, 0};
  i32x8 zz8 = {0, 0, 0, 0, 0, 0, 0, 0};
  __builtin_amdgcn_tensor_load_to_lds(g0, g1, zz4, zz4, zz8, 0);
}
#endif

// ============================================================
// Kernel 0: weight prep — convert+transpose all four projection
// matrices: W[k][n] fp32 -> Wt[n][k] bf16 (K-contiguous rows).
// ============================================================
__global__ void mha_weight_prep_kernel(const float* __restrict__ Wq,
                                       const float* __restrict__ Wk,
                                       const float* __restrict__ Wv,
                                       const float* __restrict__ Wo,
                                       __bf16* __restrict__ Wtq,
                                       __bf16* __restrict__ Wtk,
                                       __bf16* __restrict__ Wtv,
                                       __bf16* __restrict__ Wto) {
  const int z = blockIdx.z;
  const float* W = (z == 0) ? Wq : (z == 1) ? Wk : (z == 2) ? Wv : Wo;
  __bf16* Wt     = (z == 0) ? Wtq : (z == 1) ? Wtk : (z == 2) ? Wtv : Wto;

  __shared__ float tile[32][33];
  const int n0 = blockIdx.x * 32;
  const int k0 = blockIdx.y * 32;
  const int tx = threadIdx.x & 31;
  const int ty = threadIdx.x >> 5;  // 0..7

#pragma unroll
  for (int rr = 0; rr < 32; rr += 8)
    tile[ty + rr][tx] = W[(long)(k0 + ty + rr) * CDM + n0 + tx];
  __syncthreads();
#pragma unroll
  for (int rr = 0; rr < 32; rr += 8)
    Wt[(long)(n0 + ty + rr) * CDM + k0 + tx] = (__bf16)tile[tx][ty + rr];
}

// ============================================================
// Kernel 1: fused QKV projection (bf16 WMMA GEMM), bias add.
// Layouts: Q,K -> [B,H,S,64] (natural);  V -> [B,H,64,S] (transposed).
// block = 256 thr (8 waves), tile 128(M) x 64(N), K-chunk 32.
// Weight tiles arrive in LDS via the Tensor Data Mover (double-
// buffered, wave0 issues + s_wait_tensorcnt); TDM padding recreates
// the conflict-free LDSP row stride.
// ============================================================
__global__ void mha_qkv_proj_kernel(const float* __restrict__ Xq,
                                    const float* __restrict__ Xk,
                                    const float* __restrict__ Xv,
                                    const __bf16* __restrict__ Wtq, const float* __restrict__ bq,
                                    const __bf16* __restrict__ Wtk, const float* __restrict__ bk,
                                    const __bf16* __restrict__ Wtv, const float* __restrict__ bv,
                                    __bf16* __restrict__ Qbf,
                                    __bf16* __restrict__ Kbf,
                                    __bf16* __restrict__ Vtbf) {
  const int z = blockIdx.z;
  const float*  X    = (z == 0) ? Xq  : (z == 1) ? Xk  : Xv;
  const __bf16* Wt   = (z == 0) ? Wtq : (z == 1) ? Wtk : Wtv;
  const float*  bias = (z == 0) ? bq  : (z == 1) ? bk  : bv;

  const int m0   = blockIdx.x * 128;
  const int n0   = blockIdx.y * 64;
  const int tid  = threadIdx.x;
  const int wave = tid >> 5;
  const int lane = tid & 31;
  const int ln   = lane & 15;
  const int hi   = lane >> 4;

  v8f acc[4];
#pragma unroll
  for (int nt = 0; nt < 4; ++nt) acc[nt] = v8f_zero();

  const int mRow = m0 + wave * 16 + ln;
  const float* arow = X + (long)mRow * CDM;

#if USE_TDM
  __shared__ __align__(16) __bf16 ldsWt[2][64 * LDSP];
  if (wave == 0) {
    tdm_load_tile_2d(Wt + (long)n0 * CDM, (unsigned)(unsigned long long)&ldsWt[0][0],
                     CDM, CDM, 32, 64, CDM);
  }
  int p = 0;
  for (int k0 = 0; k0 < CDM; k0 += 32) {
    if (wave == 0) __builtin_amdgcn_s_wait_tensorcnt(0);
    __syncthreads();
    if (wave == 0 && k0 + 32 < CDM) {
      tdm_load_tile_2d(Wt + (long)n0 * CDM + k0 + 32,
                       (unsigned)(unsigned long long)&ldsWt[1 - p][0],
                       CDM, CDM, 32, 64, CDM);
    }
    v16bf afrag = load_frag16_f32(arow + k0, hi);
#pragma unroll
    for (int nt = 0; nt < 4; ++nt) {
      v16bf bfrag = load_frag16(&ldsWt[p][(nt * 16 + ln) * LDSP], hi);
      acc[nt] = wmma_bf16(afrag, bfrag, acc[nt]);
    }
    p ^= 1;
  }
#else
  for (int k0 = 0; k0 < CDM; k0 += 32) {
    v16bf afrag = load_frag16_f32(arow + k0, hi);
#pragma unroll
    for (int nt = 0; nt < 4; ++nt) {
      v16bf bfrag = load_frag16(Wt + (long)(n0 + nt * 16 + ln) * CDM + k0, hi);
      acc[nt] = wmma_bf16(afrag, bfrag, acc[nt]);
    }
  }
#endif

  // epilogue: bias + convert + scatter to attention layouts
#pragma unroll
  for (int nt = 0; nt < 4; ++nt) {
    const int n     = n0 + nt * 16 + ln;
    const float bv_ = bias[n];
    const int hcol  = n >> 6;   // head
    const int hd    = n & 63;   // dim within head
#pragma unroll
    for (int r = 0; r < 8; ++r) {
      const int m  = m0 + wave * 16 + r + 8 * hi;
      const int bb = m >> 11;          // m / S  (S = 2048)
      const int s  = m & (CS - 1);
      const __bf16 val = (__bf16)(acc[nt][r] + bv_);
      if (z == 2) {
        Vtbf[((long)(bb * CH + hcol) * CHD + hd) * CS + s] = val;   // transposed
      } else {
        __bf16* dst = (z == 0) ? Qbf : Kbf;
        dst[((long)(bb * CH + hcol) * CS + s) * CHD + hd] = val;    // natural
      }
    }
  }
}

// ============================================================
// Kernel 2: per-key-column softmax stats over the QUERY axis.
// For each (b,h,k): m[k] = max_q s[q,k], l[k] = sum_q exp(s-m).
// Online accumulation over q-tiles of 16; each wave owns 16 key cols.
// ============================================================
__global__ void mha_col_stats_kernel(const __bf16* __restrict__ Qbf,
                                     const __bf16* __restrict__ Kbf,
                                     float* __restrict__ mArr,
                                     float* __restrict__ lArr) {
  const int b = blockIdx.z, h = blockIdx.y;
  const int tid  = threadIdx.x;
  const int wave = tid >> 5;
  const int lane = tid & 31;
  const int ln   = lane & 15;
  const int hi   = lane >> 4;
  const int kbase = blockIdx.x * 128 + wave * 16;

  const __bf16* qBase = Qbf + (long)(b * CH + h) * CS * CHD;
  const __bf16* kBase = Kbf + (long)(b * CH + h) * CS * CHD;

  // B frags: lane's column = key row of K (contiguous in d)
  const __bf16* kRow = kBase + (long)(kbase + ln) * CHD;
  const v16bf bf0 = load_frag16(kRow, hi);        // d 0..31
  const v16bf bf1 = load_frag16(kRow + 32, hi);   // d 32..63

  float run_m = -3.0e38f, run_l = 0.f;

  for (int qt = 0; qt < CS; qt += 16) {
    const __bf16* arow = qBase + (long)(qt + ln) * CHD;
    v16bf a0 = load_frag16(arow, hi);
    v16bf a1 = load_frag16(arow + 32, hi);
    v8f s = v8f_zero();
    s = wmma_bf16(a0, bf0, s);
    s = wmma_bf16(a1, bf1, s);

    // reduce over the 16 q-rows of this tile (8 regs + cross-half shuffle)
    float tmax = s[0];
#pragma unroll
    for (int r = 1; r < 8; ++r) tmax = fmaxf(tmax, s[r]);
    tmax = fmaxf(tmax, __shfl_xor(tmax, 16, 32));

    float tsum = 0.f;
#pragma unroll
    for (int r = 0; r < 8; ++r) tsum += __expf(s[r] - tmax);
    tsum += __shfl_xor(tsum, 16, 32);

    const float nm = fmaxf(run_m, tmax);
    run_l = run_l * __expf(run_m - nm) + tsum * __expf(tmax - nm);
    run_m = nm;
  }

  if (hi == 0) {
    const long idx = (long)(b * CH + h) * CS + kbase + ln;
    mArr[idx] = run_m;
    lArr[idx] = run_l;
  }
}

// ============================================================
// Kernel 3: out[q,d] = sum_k exp(s[q,k]-m[k]) / (l[k]*8) * v[k,d].
// Scores via WMMA (B-frags from K natural rows), P tile staged in
// LDS (padded rows) to convert C-layout -> A-fragment, then P x V
// with B-frags from V transposed. ctx -> [B,S,H*64] bf16.
// ============================================================
__global__ void mha_attn_out_kernel(const __bf16* __restrict__ Qbf,
                                    const __bf16* __restrict__ Kbf,
                                    const __bf16* __restrict__ Vtbf,
                                    const float* __restrict__ mArr,
                                    const float* __restrict__ lArr,
                                    __bf16* __restrict__ Ctx) {
  const int b = blockIdx.z, h = blockIdx.y;
  const int tid  = threadIdx.x;
  const int wave = tid >> 5;
  const int lane = tid & 31;
  const int ln   = lane & 15;
  const int hi   = lane >> 4;
  const int qbase = blockIdx.x * 128 + wave * 16;

  const __bf16* qBase  = Qbf  + (long)(b * CH + h) * CS * CHD;
  const __bf16* kBase  = Kbf  + (long)(b * CH + h) * CS * CHD;
  const __bf16* vtBase = Vtbf + (long)(b * CH + h) * CHD * CS;
  const float* mRow = mArr + (long)(b * CH + h) * CS;
  const float* lRow = lArr + (long)(b * CH + h) * CS;

  // per-wave P staging: 16 rows x 32 k, padded row stride 40
  __shared__ __align__(16) __bf16 pTile[8][16 * LDSP];
  __bf16* myP = pTile[wave];

  const __bf16* arow = qBase + (long)(qbase + ln) * CHD;
  const v16bf a0 = load_frag16(arow, hi);
  const v16bf a1 = load_frag16(arow + 32, hi);

  v8f acc[4];
#pragma unroll
  for (int nt = 0; nt < 4; ++nt) acc[nt] = v8f_zero();

  for (int kc = 0; kc < CS; kc += 32) {
    // prefetch next K chunk (global_prefetch path); lanes cover 32 rows
    if (kc + 32 < CS) __builtin_prefetch(kBase + (long)(kc + 32 + lane) * CHD, 0, 0);

    // scores for two 16-column subtiles, exp-normalize, stage to LDS
#pragma unroll
    for (int ks = 0; ks < 2; ++ks) {
      const int kcol0 = kc + ks * 16;
      const __bf16* kRow = kBase + (long)(kcol0 + ln) * CHD;
      v16bf bk0 = load_frag16(kRow, hi);
      v16bf bk1 = load_frag16(kRow + 32, hi);
      v8f s = v8f_zero();
      s = wmma_bf16(a0, bk0, s);
      s = wmma_bf16(a1, bk1, s);

      const int kcol = kcol0 + ln;
      const float mK  = mRow[kcol];
      const float inv = 1.0f / (lRow[kcol] * 8.0f);  // fold 1/sqrt(64)
#pragma unroll
      for (int r = 0; r < 8; ++r) {
        const float p = __expf(s[r] - mK) * inv;
        myP[(r + 8 * hi) * LDSP + ks * 16 + ln] = (__bf16)p;
      }
    }
    __syncthreads();

    // P (16x32) as an A fragment from LDS; B = Vt columns (K-contiguous)
    v16bf ap = load_frag16(myP + ln * LDSP, hi);
#pragma unroll
    for (int nt = 0; nt < 4; ++nt) {
      v16bf bvf = load_frag16(vtBase + (long)(nt * 16 + ln) * CS + kc, hi);
      acc[nt] = wmma_bf16(ap, bvf, acc[nt]);
    }
    __syncthreads();
  }

  // write ctx in [B, S, H*64] (merged heads) as bf16 for the final GEMM
#pragma unroll
  for (int nt = 0; nt < 4; ++nt) {
#pragma unroll
    for (int r = 0; r < 8; ++r) {
      const int q = qbase + r + 8 * hi;
      Ctx[((long)b * CS + q) * CDM + h * CHD + nt * 16 + ln] = (__bf16)acc[nt][r];
    }
  }
}

// ============================================================
// Kernel 4: output projection  Out = Ctx(bf16) @ Wo + bo  (fp32 out)
// Same tiling as kernel 1; A already bf16, TDM-staged Wto tiles.
// ============================================================
__global__ void mha_out_proj_kernel(const __bf16* __restrict__ Ctx,
                                    const __bf16* __restrict__ Wto,
                                    const float* __restrict__ bo,
                                    float* __restrict__ Out) {
  const int m0   = blockIdx.x * 128;
  const int n0   = blockIdx.y * 64;
  const int tid  = threadIdx.x;
  const int wave = tid >> 5;
  const int lane = tid & 31;
  const int ln   = lane & 15;
  const int hi   = lane >> 4;

  v8f acc[4];
#pragma unroll
  for (int nt = 0; nt < 4; ++nt) acc[nt] = v8f_zero();

  const int mRow = m0 + wave * 16 + ln;
  const __bf16* arow = Ctx + (long)mRow * CDM;

#if USE_TDM
  __shared__ __align__(16) __bf16 ldsWt[2][64 * LDSP];
  if (wave == 0) {
    tdm_load_tile_2d(Wto + (long)n0 * CDM, (unsigned)(unsigned long long)&ldsWt[0][0],
                     CDM, CDM, 32, 64, CDM);
  }
  int p = 0;
  for (int k0 = 0; k0 < CDM; k0 += 32) {
    if (wave == 0) __builtin_amdgcn_s_wait_tensorcnt(0);
    __syncthreads();
    if (wave == 0 && k0 + 32 < CDM) {
      tdm_load_tile_2d(Wto + (long)n0 * CDM + k0 + 32,
                       (unsigned)(unsigned long long)&ldsWt[1 - p][0],
                       CDM, CDM, 32, 64, CDM);
    }
    v16bf afrag = load_frag16(arow + k0, hi);
#pragma unroll
    for (int nt = 0; nt < 4; ++nt) {
      v16bf bfrag = load_frag16(&ldsWt[p][(nt * 16 + ln) * LDSP], hi);
      acc[nt] = wmma_bf16(afrag, bfrag, acc[nt]);
    }
    p ^= 1;
  }
#else
  for (int k0 = 0; k0 < CDM; k0 += 32) {
    v16bf afrag = load_frag16(arow + k0, hi);
#pragma unroll
    for (int nt = 0; nt < 4; ++nt) {
      v16bf bfrag = load_frag16(Wto + (long)(n0 + nt * 16 + ln) * CDM + k0, hi);
      acc[nt] = wmma_bf16(afrag, bfrag, acc[nt]);
    }
  }
#endif

#pragma unroll
  for (int nt = 0; nt < 4; ++nt) {
    const int n = n0 + nt * 16 + ln;
    const float bv_ = bo[n];
#pragma unroll
    for (int r = 0; r < 8; ++r) {
      const int m = m0 + wave * 16 + r + 8 * hi;
      Out[(long)m * CDM + n] = acc[nt][r] + bv_;
    }
  }
}

// ============================================================
extern "C" void kernel_launch(void* const* d_in, const int* in_sizes, int n_in,
                              void* d_out, int out_size, void* d_ws, size_t ws_size,
                              hipStream_t stream) {
  const float* Xq = (const float*)d_in[0];
  const float* Xk = (const float*)d_in[1];
  const float* Xv = (const float*)d_in[2];
  const float* Wq = (const float*)d_in[3];
  const float* bq = (const float*)d_in[4];
  const float* Wk = (const float*)d_in[5];
  const float* bk = (const float*)d_in[6];
  const float* Wv = (const float*)d_in[7];
  const float* bv = (const float*)d_in[8];
  const float* Wo = (const float*)d_in[9];
  const float* bo = (const float*)d_in[10];

  char* ws = (char*)d_ws;
  const size_t qkvElems = (size_t)CB * CH * CS * CHD;  // 8,388,608
  const size_t wElems   = (size_t)CDM * CDM;           // 1,048,576
  __bf16* Qbf  = (__bf16*)(ws);
  __bf16* Kbf  = (__bf16*)(ws + 1 * qkvElems * 2);
  __bf16* Vtbf = (__bf16*)(ws + 2 * qkvElems * 2);
  __bf16* Ctx  = (__bf16*)(ws + 3 * qkvElems * 2);
  float*  mArr = (float*) (ws + 4 * qkvElems * 2);
  float*  lArr = mArr + (size_t)CB * CH * CS;
  char*   wsW  = (char*)(lArr + (size_t)CB * CH * CS);
  __bf16* Wtq  = (__bf16*)(wsW);
  __bf16* Wtk  = (__bf16*)(wsW + 1 * wElems * 2);
  __bf16* Wtv  = (__bf16*)(wsW + 2 * wElems * 2);
  __bf16* Wto  = (__bf16*)(wsW + 3 * wElems * 2);

  dim3 blk(256);
  // 0) weight convert+transpose: 32x32 tiles, z selects matrix
  mha_weight_prep_kernel<<<dim3(32, 32, 4), blk, 0, stream>>>(
      Wq, Wk, Wv, Wo, Wtq, Wtk, Wtv, Wto);
  // 1) M=8192 rows in tiles of 128; N=1024 in tiles of 64; z selects Q/K/V
  mha_qkv_proj_kernel<<<dim3(64, 16, 3), blk, 0, stream>>>(
      Xq, Xk, Xv, Wtq, bq, Wtk, bk, Wtv, bv, Qbf, Kbf, Vtbf);
  // 2) 2048 key columns in tiles of 128 per (h,b)
  mha_col_stats_kernel<<<dim3(16, CH, CB), blk, 0, stream>>>(Qbf, Kbf, mArr, lArr);
  // 3) 2048 query rows in tiles of 128 per (h,b)
  mha_attn_out_kernel<<<dim3(16, CH, CB), blk, 0, stream>>>(Qbf, Kbf, Vtbf, mArr,
                                                            lArr, Ctx);
  // 4) output projection
  mha_out_proj_kernel<<<dim3(64, 16, 1), blk, 0, stream>>>(Ctx, Wto, bo,
                                                           (float*)d_out);
}
